// LinkPred_27092653703582
// MI455X (gfx1250) — compile-verified, matrix operand order checked by reading, not run on hardware
//
#include <hip/hip_runtime.h>
#include <hip/hip_bf16.h>

// ---------------- problem constants ----------------
#define NN    8192
#define EE    16384
#define DD    256
#define NNZN  262144
#define PP    32
#define CAP   (1u << 20)      // hash capacity (4x NNZ)
#define CAPM  (CAP - 1u)
#define EMPTYK 0xFFFFFFFFu

typedef __attribute__((ext_vector_type(2)))  float    v2f;
typedef __attribute__((ext_vector_type(4)))  float    v4f;
typedef __attribute__((ext_vector_type(8)))  float    v8f;
typedef __attribute__((ext_vector_type(16))) _Float16 v16h;

// scalar -> P -> scalar MLP with ReLU hidden layer
__device__ __forceinline__ float mlp1(float x, const float* w1, const float* b1,
                                      const float* w2, float b2) {
    float acc = b2;
#pragma unroll
    for (int j = 0; j < PP; ++j) {
        float t = fmaf(x, w1[j], b1[j]);
        t = t > 0.f ? t : 0.f;
        acc = fmaf(t, w2[j], acc);
    }
    return acc;
}

__device__ __forceinline__ float sigm(float x) { return 1.f / (1.f + __expf(-x)); }

// ---------------- K0: init hash table + accumulators ----------------
__global__ void lp_init(unsigned* keys, float* vals, float* node_in, float* Sacc) {
    int i = blockIdx.x * blockDim.x + threadIdx.x;
    if (i < (int)CAP) { keys[i] = EMPTYK; vals[i] = 0.f; }
    if (i < NN)       { node_in[i] = 0.f; Sacc[i] = 0.f; }
}

// ---------------- K1: hash-insert adj entries; edge MLP + segment sum ----------------
__global__ void lp_scatter(const int* __restrict__ row, const int* __restrict__ col,
                           const float* __restrict__ values,
                           const float* __restrict__ we1, const float* __restrict__ be1,
                           const float* __restrict__ we2, const float* __restrict__ be2,
                           unsigned* keys, float* vals, float* node_in) {
    int i = blockIdx.x * blockDim.x + threadIdx.x;
    if (i >= NNZN) return;
    int r = row[i], c = col[i];
    float v = values[i];
    unsigned key = ((unsigned)r << 13) | (unsigned)c;   // r,c < 8192 -> key < 2^26
    unsigned h = (key * 2654435761u >> 12) & CAPM;
    for (;;) {
        unsigned prev = atomicCAS(&keys[h], EMPTYK, key);
        if (prev == EMPTYK || prev == key) { atomicAdd(&vals[h], v); break; }
        h = (h + 1u) & CAPM;
    }
    float ef = mlp1(v, we1, be1, we2, be2[0]);
    atomicAdd(&node_in[c], ef);
}

// ---------------- K2: node MLP -> struct^2 ----------------
__global__ void lp_struct(const float* __restrict__ node_in,
                          const float* __restrict__ wn1, const float* __restrict__ bn1,
                          const float* __restrict__ wn2, const float* __restrict__ bn2,
                          float* struct2) {
    int u = blockIdx.x * blockDim.x + threadIdx.x;
    if (u >= NN) return;
    float s = mlp1(node_in[u], wn1, bn1, wn2, bn2[0]);
    struct2[u] = s * s;
}

// ---------------- K3: sweep hash table -> S[u] = sum_c adj[u,c]^2 * struct2[c] ----------------
__global__ void lp_agg(const unsigned* __restrict__ keys, const float* __restrict__ vals,
                       const float* __restrict__ struct2, float* Sacc) {
    int i = blockIdx.x * blockDim.x + threadIdx.x;
    if (i >= (int)CAP) return;
    unsigned key = keys[i];
    if (key == EMPTYK) return;
    int r = (int)(key >> 13);
    int c = (int)(key & 8191u);
    float v = vals[i];
    atomicAdd(&Sacc[r], v * v * struct2[c]);
}

// ---------------- K4: WMMA batched dot products + final combine ----------------
// 256 threads = 8 waves; each wave computes 16 edge dot products via WMMA.
// E = 16384 = 128 blocks * 8 waves * 16 edges exactly (no tail -> EXEC all-ones at WMMA).
// Each lane loads one float4 (b128) per operand per 8-K chunk and feeds two WMMAs:
//   step 1 slot map: K = k0 + 4*half + v     -> {k0, k0+1, k0+4, k0+5}
//   step 2 slot map: K = k0 + 4*half + 2 + v -> {k0+2, k0+3, k0+6, k0+7}
// A and B use identical slot maps, so the accumulated diagonal is the exact dot.
__global__ void lp_final(const int* __restrict__ edges, const float* __restrict__ H,
                         const float* __restrict__ Sacc,
                         const float* __restrict__ wp1, const float* __restrict__ bp1,
                         const float* __restrict__ wp2, const float* __restrict__ bp2,
                         const float* __restrict__ alpha, float* __restrict__ out) {
    const int lane = threadIdx.x & 31;
    const int wave = threadIdx.x >> 5;
    const int eb   = blockIdx.x * 128 + wave * 16;   // first edge of this wave's tile
    const int m    = lane & 15;
    const int half = lane >> 4;
    const int e    = eb + m;
    const int s    = edges[2 * e + 0];
    const int d    = edges[2 * e + 1];
    const float* __restrict__ Ar = H + (size_t)s * DD + 4 * half;  // A: src row (M = m)
    const float* __restrict__ Br = H + (size_t)d * DD + 4 * half;  // B: dst row (N = m)

    v8f c = {};
#if __has_builtin(__builtin_amdgcn_wmma_f32_16x16x4_f32)
#pragma unroll 4
    for (int k0 = 0; k0 < DD; k0 += 8) {
        v4f a4 = *(const v4f*)(Ar + k0);
        v4f b4 = *(const v4f*)(Br + k0);
        v2f a0, b0, a1, b1;
        a0.x = a4.x; a0.y = a4.y;  b0.x = b4.x; b0.y = b4.y;
        a1.x = a4.z; a1.y = a4.w;  b1.x = b4.z; b1.y = b4.w;
        c = __builtin_amdgcn_wmma_f32_16x16x4_f32(false, a0, false, b0,
                                                  (short)0, c, false, false);
        c = __builtin_amdgcn_wmma_f32_16x16x4_f32(false, a1, false, b1,
                                                  (short)0, c, false, false);
    }
#else
    // f16 fallback: K=32 per step; slot map K(t,half) = t + 8*half + (t>=8 ? 8 : 0),
    // identical for A and B.
#pragma unroll 2
    for (int k0 = 0; k0 < DD; k0 += 32) {
        v16h a, b;
#pragma unroll
        for (int t = 0; t < 16; ++t) {
            int K = k0 + t + 4 * half + (t >= 8 ? 8 : 0) - 4 * half + 8 * half; // = k0+t+8*half(+8)
            a[t] = (_Float16)Ar[K - 4 * half];
            b[t] = (_Float16)Br[K - 4 * half];
        }
        c = __builtin_amdgcn_wmma_f32_16x16x32_f16(false, a, false, b,
                                                   (short)0, c, false, false);
    }
#endif

    // Diagonal extraction (16x16 f32 C layout): edge m in lane m (m<8, VGPR m)
    // and lane m+16 (m>=8, VGPR m-8). Lanes 8..23 hold no diagonal.
    bool have = (lane < 8) || (lane >= 24);
    int  ridx = lane & 7;
    float dotv = 0.f;
#pragma unroll
    for (int i = 0; i < 8; ++i)
        if (ridx == i) dotv = c[i];

    if (have) {
        float h_link = sigm(dotv);
        float agg    = Sacc[s];                 // agg[e] = S[src_e]
        float z_link = sigm(mlp1(agg, wp1, bp1, wp2, bp2[0]));
        float a0 = alpha[0], a1 = alpha[1];
        float mx = fmaxf(a0, a1);
        float e0 = __expf(a0 - mx), e1 = __expf(a1 - mx);
        float inv = 1.f / (e0 + e1);
        out[e] = (e0 * inv) * z_link + (e1 * inv) * h_link + 1e-15f;
    }
}

// ---------------- launcher ----------------
extern "C" void kernel_launch(void* const* d_in, const int* in_sizes, int n_in,
                              void* d_out, int out_size, void* d_ws, size_t ws_size,
                              hipStream_t stream) {
    const int*   edges  = (const int*)  d_in[0];
    const int*   row    = (const int*)  d_in[1];
    const int*   col    = (const int*)  d_in[2];
    const float* values = (const float*)d_in[3];
    const float* H      = (const float*)d_in[4];
    const float* we1    = (const float*)d_in[5];
    const float* be1    = (const float*)d_in[6];
    const float* we2    = (const float*)d_in[7];
    const float* be2    = (const float*)d_in[8];
    const float* wn1    = (const float*)d_in[9];
    const float* bn1    = (const float*)d_in[10];
    const float* wn2    = (const float*)d_in[11];
    const float* bn2    = (const float*)d_in[12];
    const float* wp1    = (const float*)d_in[13];
    const float* bp1    = (const float*)d_in[14];
    const float* wp2    = (const float*)d_in[15];
    const float* bp2    = (const float*)d_in[16];
    const float* alpha  = (const float*)d_in[17];
    float* out = (float*)d_out;

    char* w = (char*)d_ws;
    unsigned* keys    = (unsigned*)(w);                          // 4 MB
    float*    vals    = (float*)   (w + (size_t)CAP * 4);        // 4 MB
    float*    node_in = (float*)   (w + (size_t)CAP * 8);        // 32 KB
    float*    struct2 = (float*)   (w + (size_t)CAP * 8 + NN * 4);
    float*    Sacc    = (float*)   (w + (size_t)CAP * 8 + NN * 8);

    lp_init   <<<CAP / 256, 256, 0, stream>>>(keys, vals, node_in, Sacc);
    lp_scatter<<<NNZN / 256, 256, 0, stream>>>(row, col, values,
                                               we1, be1, we2, be2,
                                               keys, vals, node_in);
    lp_struct <<<NN / 256, 256, 0, stream>>>(node_in, wn1, bn1, wn2, bn2, struct2);
    lp_agg    <<<CAP / 256, 256, 0, stream>>>(keys, vals, struct2, Sacc);
    lp_final  <<<EE / 128, 256, 0, stream>>>(edges, H, Sacc,
                                             wp1, bp1, wp2, bp2, alpha, out);
    (void)in_sizes; (void)n_in; (void)out_size; (void)ws_size;
}
